// OneToManyDeepRNN_6399501271111
// MI455X (gfx1250) — matrix operand order repaired and checked by reading, not compiled, not run on hardware
//
#include <hip/hip_runtime.h>
#include <math.h>

// ---------------------------------------------------------------------------
// OneToManyDeepRNN for MI455X (gfx1250, wave32, WMMA).
// Carries freeze after step 0 => steps 1..31 batch into M=1984 GEMMs;
// r[l] = hs1[l]@Wh[l] + b[l] precomputed once per layer.
// bf16 WMMA GEMM (f32 accumulate). f32->bf16 via round-half-up + v_perm_b32
// packing (3 VALU per pair). Double-buffered LDS, 64x256 block tile,
// 32x64 wave tile (8 WMMA/iter), global_prefetch of the weight stream.
// ---------------------------------------------------------------------------

#define B_ 64
#define T_ 32
#define D_ 512
#define H_ 1024
#define L_ 4
#define G4H 4096

typedef __attribute__((ext_vector_type(16))) __bf16 v16bf;
typedef __attribute__((ext_vector_type(2)))  __bf16 v2bf;
typedef __attribute__((ext_vector_type(8)))  float  v8f;

union Frag {
    uint4 q[2];
    v16bf v;
};

// Pack two f32 into one dword of bf16 (round-half-up).
__device__ __forceinline__ unsigned int pack_bf16(float x, float y) {
#if __has_builtin(__builtin_amdgcn_cvt_pk_bf16_f32)
    v2bf p = __builtin_amdgcn_cvt_pk_bf16_f32(x, y);
    return __builtin_bit_cast(unsigned int, p);
#elif __has_builtin(__builtin_amdgcn_perm)
    unsigned int ux = __float_as_uint(x) + 0x8000u;
    unsigned int uy = __float_as_uint(y) + 0x8000u;
    // result bytes (hi..lo) = { uy.b3, uy.b2, ux.b3, ux.b2 }
    return __builtin_amdgcn_perm(uy, ux, 0x07060302u);
#else
    unsigned int ux = __float_as_uint(x) + 0x8000u;
    unsigned int uy = __float_as_uint(y) + 0x8000u;
    return (ux >> 16) | (uy & 0xFFFF0000u);
#endif
}

// ---------------------------------------------------------------------------
// out = A(f32)[M,K] * Bm(f32)[K,N] (+bias[n]) (+res[m%64,n])
//   A_MODE 0: A row gm at A + gm*K             (row-major, lda==K)
//   A_MODE 1: A row gm = x[gm%64, a_t0+gm/64, :]
//   O_MODE 0: out[gm*N + n]
//   O_MODE 1: out[(((o_t0+gm/64)*L + o_l)*64 + gm%64)*1024 + n]
// M%64==0, N%256==0, K%32==0. Block 256 (8 waves), tile 64x256, BK=32,
// wave tile 32x64 (2x4 WMMA 16x16x32 bf16), double-buffered LDS.
// ---------------------------------------------------------------------------
template<int A_MODE, int O_MODE, bool HAS_BIAS, bool HAS_RES>
__global__ void __launch_bounds__(256)
wmma_gemm_bf16(const float* __restrict__ A, const float* __restrict__ Bm,
               const float* __restrict__ bias, const float* __restrict__ res,
               float* __restrict__ out,
               int M, int N, int K, int a_t0, int o_t0, int o_l)
{
    // Row stride 40 ushorts = 80B (16B-aligned, conflict-free b128 frag loads).
    __shared__ __align__(16) unsigned short As[2][64 * 40];    //  10 KB
    __shared__ __align__(16) unsigned short Bs[2][256 * 40];   //  40 KB

    const int tid  = threadIdx.x;
    const int lane = tid & 31;
    const int wave = tid >> 5;
    const int wm   = wave >> 2;        // 0..1 : 32-row band
    const int wn   = wave & 3;         // 0..3 : 64-col band
    const int half = lane >> 4;
    const int lr   = lane & 15;

    const int bm = blockIdx.y;
    const int bn = blockIdx.x;

    // ---- loop-invariant staging addresses ----
    // A: 64x32 tile, 2 float4 per thread
    const float* aptr[2];
    int asIdx[2];
    #pragma unroll
    for (int i = 0; i < 2; ++i) {
        int q   = tid + i * 256;
        int row = q >> 3;
        int c4  = (q & 7) << 2;
        int gm  = bm * 64 + row;
        size_t arow = (A_MODE == 0)
                    ? (size_t)gm * K
                    : ((size_t)(gm & 63) * T_ + (size_t)(a_t0 + (gm >> 6))) * D_;
        aptr[i]  = A + arow + c4;
        asIdx[i] = row * 40 + c4;
    }
    // B: 32x256 tile, 8 float4 per thread, stored transposed [n][k]
    const float* bptr[8];
    int bsIdx[8];
    #pragma unroll
    for (int i = 0; i < 8; ++i) {
        int q    = tid + i * 256;
        int krow = q >> 6;             // 64 float4 per 256-wide row
        int n4   = (q & 63) << 2;
        bptr[i]  = Bm + (size_t)krow * N + bn * 256 + n4;
        bsIdx[i] = n4 * 40 + krow;
    }

    // ---- loop-invariant fragment offsets ----
    const int aoff[2] = { (wm * 32 + lr) * 40 + half * 8,
                          (wm * 32 + 16 + lr) * 40 + half * 8 };
    int boff[4];
    #pragma unroll
    for (int ni = 0; ni < 4; ++ni)
        boff[ni] = (wn * 64 + ni * 16 + lr) * 40 + half * 8;

    v8f acc[2][4];
    #pragma unroll
    for (int i = 0; i < 2; ++i)
        #pragma unroll
        for (int j = 0; j < 4; ++j)
            acc[i][j] = (v8f)0.0f;

    float4 ra[2], rb[8];

    // prologue: fetch tile 0 and stage into buffer 0
    #pragma unroll
    for (int i = 0; i < 2; ++i) ra[i] = *(const float4*)(aptr[i]);
    #pragma unroll
    for (int i = 0; i < 8; ++i) rb[i] = *(const float4*)(bptr[i]);
    #pragma unroll
    for (int i = 0; i < 2; ++i) {
        uint2 p;
        p.x = pack_bf16(ra[i].x, ra[i].y);
        p.y = pack_bf16(ra[i].z, ra[i].w);
        *(uint2*)&As[0][asIdx[i]] = p;
    }
    #pragma unroll
    for (int i = 0; i < 8; ++i) {
        unsigned int p01 = pack_bf16(rb[i].x, rb[i].y);
        unsigned int p23 = pack_bf16(rb[i].z, rb[i].w);
        Bs[0][bsIdx[i]      ] = (unsigned short)(p01);
        Bs[0][bsIdx[i] + 40 ] = (unsigned short)(p01 >> 16);
        Bs[0][bsIdx[i] + 80 ] = (unsigned short)(p23);
        Bs[0][bsIdx[i] + 120] = (unsigned short)(p23 >> 16);
    }

    const int nsteps = K >> 5;
    for (int s = 0; s < nsteps; ++s) {
        __syncthreads();
        const int cur = s & 1;

        // issue next tile's global loads (latency hidden under WMMA)
        if (s + 1 < nsteps) {
            const int k0 = (s + 1) << 5;
            #pragma unroll
            for (int i = 0; i < 2; ++i) ra[i] = *(const float4*)(aptr[i] + k0);
            #pragma unroll
            for (int i = 0; i < 8; ++i) rb[i] = *(const float4*)(bptr[i] + (size_t)k0 * N);
        }
        // prefetch the weight stream two steps ahead (global_prefetch_b8)
        if (s + 2 < nsteps) {
            const size_t k2 = (size_t)((s + 2) << 5) * N;
            #pragma unroll
            for (int i = 0; i < 8; i += 2)
                __builtin_prefetch(bptr[i] + k2, 0, 3);
        }

        // fragments (ISA 16-bit A/B layouts) + 2x4 WMMA
        Frag a[2], b[4];
        #pragma unroll
        for (int mi = 0; mi < 2; ++mi) {
            a[mi].q[0] = *(const uint4*)&As[cur][aoff[mi]];
            a[mi].q[1] = *(const uint4*)&As[cur][aoff[mi] + 16];
        }
        #pragma unroll
        for (int ni = 0; ni < 4; ++ni) {
            b[ni].q[0] = *(const uint4*)&Bs[cur][boff[ni]];
            b[ni].q[1] = *(const uint4*)&Bs[cur][boff[ni] + 16];
        }

        #pragma unroll
        for (int mi = 0; mi < 2; ++mi)
            #pragma unroll
            for (int ni = 0; ni < 4; ++ni)
                acc[mi][ni] = __builtin_amdgcn_wmma_f32_16x16x32_bf16(
                    false, a[mi].v, false, b[ni].v,
                    (short)0, acc[mi][ni], false, false);

        // stage next tile into the other buffer
        if (s + 1 < nsteps) {
            const int nxt = cur ^ 1;
            #pragma unroll
            for (int i = 0; i < 2; ++i) {
                uint2 p;
                p.x = pack_bf16(ra[i].x, ra[i].y);
                p.y = pack_bf16(ra[i].z, ra[i].w);
                *(uint2*)&As[nxt][asIdx[i]] = p;
            }
            #pragma unroll
            for (int i = 0; i < 8; ++i) {
                unsigned int p01 = pack_bf16(rb[i].x, rb[i].y);
                unsigned int p23 = pack_bf16(rb[i].z, rb[i].w);
                Bs[nxt][bsIdx[i]      ] = (unsigned short)(p01);
                Bs[nxt][bsIdx[i] + 40 ] = (unsigned short)(p01 >> 16);
                Bs[nxt][bsIdx[i] + 80 ] = (unsigned short)(p23);
                Bs[nxt][bsIdx[i] + 120] = (unsigned short)(p23 >> 16);
            }
        }
    }

    // ---- epilogue (templated: no branches) ----
    #pragma unroll
    for (int mi = 0; mi < 2; ++mi) {
        #pragma unroll
        for (int ni = 0; ni < 4; ++ni) {
            const int n = bn * 256 + wn * 64 + ni * 16 + lr;
            const float bn_add = HAS_BIAS ? bias[n] : 0.0f;
            #pragma unroll
            for (int v = 0; v < 8; ++v) {
                const int ml = wm * 32 + mi * 16 + v + 8 * half;
                const int gm = bm * 64 + ml;
                float val = acc[mi][ni][v] + bn_add;
                if (HAS_RES) val += res[(size_t)(gm & 63) * N + n];
                size_t off;
                if (O_MODE == 0) {
                    off = (size_t)gm * N + n;
                } else {
                    size_t t = (size_t)(o_t0 + (gm >> 6));
                    off = (((t * L_ + o_l) << 6) + (size_t)(gm & 63)) * 1024 + n;
                }
                out[off] = val;
            }
        }
    }
}

// ---------------------------------------------------------------------------
__device__ __forceinline__ float sigm(float x) { return 1.0f / (1.0f + __expf(-x)); }

__global__ void __launch_bounds__(256)
lstm_gates(const float* __restrict__ Z, const float* __restrict__ cprev,
           float* __restrict__ hout, float* __restrict__ cout, int M)
{
    int idx = blockIdx.x * blockDim.x + threadIdx.x;
    if (idx >= M * H_) return;
    int m = idx >> 10;
    int j = idx & 1023;
    const float* z = Z + (size_t)m * G4H;
    float zi = z[j];
    float zf = z[1024 + j];
    float zg = z[2048 + j];
    float zo = z[3072 + j];
    float cp = cprev[((size_t)(m & 63) << 10) + j];
    float c  = sigm(zf) * cp + sigm(zi) * tanhf(zg);
    float h  = sigm(zo) * tanhf(c);
    hout[idx] = h;
    if (cout) cout[idx] = c;
}

// ---------------------------------------------------------------------------
static inline void launch_gemm(hipStream_t s,
                               const float* A, const float* Bm,
                               const float* bias, const float* res, float* out,
                               int M, int N, int K,
                               int a_mode, int a_t0,
                               int o_mode, int o_t0, int o_l)
{
    dim3 g(N / 256, M / 64), b(256);
    if (a_mode == 0 && o_mode == 0 &&  bias && !res)
        wmma_gemm_bf16<0,0,true ,false><<<g,b,0,s>>>(A,Bm,bias,res,out,M,N,K,a_t0,o_t0,o_l);
    else if (a_mode == 1 && o_mode == 0 &&  bias && !res)
        wmma_gemm_bf16<1,0,true ,false><<<g,b,0,s>>>(A,Bm,bias,res,out,M,N,K,a_t0,o_t0,o_l);
    else if (a_mode == 0 && o_mode == 1 &&  bias && !res)
        wmma_gemm_bf16<0,1,true ,false><<<g,b,0,s>>>(A,Bm,bias,res,out,M,N,K,a_t0,o_t0,o_l);
    else if (a_mode == 1 && o_mode == 0 && !bias &&  res)
        wmma_gemm_bf16<1,0,false,true ><<<g,b,0,s>>>(A,Bm,bias,res,out,M,N,K,a_t0,o_t0,o_l);
    else if (a_mode == 0 && o_mode == 0 && !bias &&  res)
        wmma_gemm_bf16<0,0,false,true ><<<g,b,0,s>>>(A,Bm,bias,res,out,M,N,K,a_t0,o_t0,o_l);
    else
        wmma_gemm_bf16<0,0,false,false><<<g,b,0,s>>>(A,Bm,bias,res,out,M,N,K,a_t0,o_t0,o_l);
}

static inline void launch_gates(hipStream_t s, const float* Z, const float* cprev,
                                float* hout, float* cout, int M)
{
    int n = M * H_;
    lstm_gates<<<dim3((n + 255) / 256), dim3(256), 0, s>>>(Z, cprev, hout, cout, M);
}

extern "C" void kernel_launch(void* const* d_in, const int* in_sizes, int n_in,
                              void* d_out, int out_size, void* d_ws, size_t ws_size,
                              hipStream_t stream)
{
    (void)in_sizes; (void)n_in; (void)out_size; (void)ws_size;

    const float* x   = (const float*)d_in[0];   // [64,32,512]
    const float* Wx0 = (const float*)d_in[1];   // [512,4096]
    const float* Wxr = (const float*)d_in[2];   // [3,1024,4096]
    const float* Wh  = (const float*)d_in[3];   // [4,1024,4096]
    const float* bb  = (const float*)d_in[4];   // [4,4096]
    const float* Wd  = (const float*)d_in[5];   // [1024,1024]
    const float* bd  = (const float*)d_in[6];   // [1024]
    const float* c0  = (const float*)d_in[8];   // [4,64,1024]
    float* out = (float*)d_out;

    const int MB = (T_ - 1) * B_;               // 1984

    float* ws   = (float*)d_ws;
    float* Z    = ws;                           // [1984,4096]
    float* Hbig = Z    + (size_t)MB * G4H;      // [1984,1024]
    float* h1   = Hbig + (size_t)MB * H_;       // [4,64,1024] hs1
    float* cs1  = h1   + (size_t)L_ * B_ * H_;  // [4,64,1024]
    float* r    = cs1  + (size_t)L_ * B_ * H_;  // [4,64,4096]

    // ---------------- Step 0 (sequential through layers, B=64) -------------
    for (int l = 0; l < L_; ++l) {
        if (l == 0) {
            launch_gemm(stream, x, Wx0, bb, nullptr, Z,
                        B_, G4H, D_, 1, 0, 0, 0, 0);
        } else {
            launch_gemm(stream, h1 + (size_t)(l - 1) * B_ * H_,
                        Wxr + (size_t)(l - 1) * H_ * G4H, bb + (size_t)l * G4H,
                        nullptr, Z, B_, G4H, H_, 0, 0, 0, 0, 0);
        }
        launch_gates(stream, Z, c0 + (size_t)l * B_ * H_,
                     h1 + (size_t)l * B_ * H_, cs1 + (size_t)l * B_ * H_, B_);
        launch_gemm(stream, h1 + (size_t)l * B_ * H_, Wd, bd, nullptr, out,
                    B_, H_, H_, 0, 0, 1, 0, l);
    }

    // r[l] = hs1[l] @ Wh[l] + b[l]
    for (int l = 0; l < L_; ++l) {
        launch_gemm(stream, h1 + (size_t)l * B_ * H_, Wh + (size_t)l * H_ * G4H,
                    bb + (size_t)l * G4H, nullptr, r + (size_t)l * B_ * G4H,
                    B_, G4H, H_, 0, 0, 0, 0, 0);
    }

    // ---------------- Steps 1..31 batched: M = 1984 -------------------------
    for (int l = 0; l < L_; ++l) {
        if (l == 0) {
            launch_gemm(stream, x, Wx0, nullptr, r, Z,
                        MB, G4H, D_, 1, 1, 0, 0, 0);
        } else {
            launch_gemm(stream, Hbig, Wxr + (size_t)(l - 1) * H_ * G4H,
                        nullptr, r + (size_t)l * B_ * G4H, Z,
                        MB, G4H, H_, 0, 0, 0, 0, 0);
        }
        launch_gates(stream, Z, cs1 + (size_t)l * B_ * H_, Hbig, nullptr, MB);
        launch_gemm(stream, Hbig, Wd, bd, nullptr, out,
                    MB, H_, H_, 0, 0, 1, 1, l);
    }
}
